// BaseAttention_5729486373364
// MI455X (gfx1250) — matrix operand order repaired and checked by reading, not compiled, not run on hardware
//
#include <hip/hip_runtime.h>

#define DEV __device__ __forceinline__

typedef __attribute__((ext_vector_type(16))) __bf16 bf16x16;
typedef __attribute__((ext_vector_type(8)))  float  f32x8;

#define BATCH 4
#define SEQ   4096
#define DM    1024
#define NH    16
#define HD    64
#define TOK   (BATCH * SEQ)   // 16384

struct Frag { uint4 lo, hi; };   // 32 bytes = 16 bf16 = one WMMA A/B operand

DEV unsigned short f2bf(float f) {
  unsigned int u = __float_as_uint(f);
  unsigned int r = (u + 0x7FFFu + ((u >> 16) & 1u)) >> 16;  // RNE
  return (unsigned short)r;
}
DEV float bf2f(unsigned short h) {
  return __uint_as_float(((unsigned int)h) << 16);
}
DEV float eluf(float x) { return x > 0.f ? x : (__expf(x) - 1.f); }

DEV f32x8 zero8() { f32x8 z = {0.f,0.f,0.f,0.f,0.f,0.f,0.f,0.f}; return z; }

DEV f32x8 wmma_bf16(const Frag& a, const Frag& b, f32x8 c) {
  return __builtin_amdgcn_wmma_f32_16x16x32_bf16(
      false, __builtin_bit_cast(bf16x16, a),
      false, __builtin_bit_cast(bf16x16, b),
      (short)0, c, false, false);
}

// ---- CDNA5 async global->LDS copy (ASYNCcnt-tracked, per ISA 08_async_tensor) ----
DEV unsigned lds_off(const void* p) {           // flat LDS addr: low 32 bits = LDS byte offset
  return (unsigned)(size_t)p;
}
DEV void async_b128(unsigned lds, const void* gptr) {
  asm volatile("global_load_async_to_lds_b128 %0, %1, off"
               :: "v"(lds), "v"(gptr) : "memory");
}
template<int N> DEV void wait_async() {
#if __has_builtin(__builtin_amdgcn_s_wait_asynccnt)
  __builtin_amdgcn_s_wait_asynccnt(N);
#else
  asm volatile("s_wait_asynccnt %0" :: "n"(N) : "memory");
#endif
}

// ---------------- pack kernels ----------------

__global__ __launch_bounds__(256) void k_pack_x(const float* __restrict__ x,
                                                unsigned short* __restrict__ xb) {
  size_t i = ((size_t)blockIdx.x * 256 + threadIdx.x) * 4;
  float4 v = *(const float4*)(x + i);
  unsigned int p0 = (unsigned int)f2bf(v.x) | ((unsigned int)f2bf(v.y) << 16);
  unsigned int p1 = (unsigned int)f2bf(v.z) | ((unsigned int)f2bf(v.w) << 16);
  *(uint2*)(xb + i) = make_uint2(p0, p1);
}

// Wt[n][k] = bf16(W[k][n]) for 4 weight matrices (blockIdx.y selects matrix)
__global__ __launch_bounds__(256) void k_pack_wt(const float* __restrict__ W0,
                                                 const float* __restrict__ W1,
                                                 const float* __restrict__ W2,
                                                 const float* __restrict__ W3,
                                                 unsigned short* __restrict__ T) {
  const float* W = (blockIdx.y == 0) ? W0 : (blockIdx.y == 1) ? W1
                 : (blockIdx.y == 2) ? W2 : W3;
  unsigned short* out = T + (size_t)blockIdx.y * DM * DM;
  int idx = blockIdx.x * 256 + threadIdx.x;   // idx = n*1024 + k
  int n = idx >> 10, k = idx & 1023;
  out[idx] = f2bf(W[(size_t)k * DM + n]);
}

// ---------------- main GEMM: C[M x 1024] = A[M x 1024] @ W + epilogue ----------------
// MODE 0: +bq, ELU, write Qb row-major bf16
// MODE 1: +bk, ELU, write Kt transposed bf16  [ (b*NH+h)*HD + d ][ n ]
// MODE 2: +bv,      write Vt transposed bf16
// MODE 3: +bo, +x residual, write f32 row-major (d_out)
template<int MODE>
__global__ __launch_bounds__(256)
void k_gemm(const unsigned short* __restrict__ A,
            const unsigned short* __restrict__ Bt,   // Bt[n][k]
            const float* __restrict__ bias,
            const float* __restrict__ xres,
            unsigned short* __restrict__ outb,
            float* __restrict__ outf) {
  __shared__ __align__(16) unsigned short As[2][128 * 32];
  __shared__ __align__(16) unsigned short Bs[2][128 * 32];
  const int tid  = threadIdx.x;
  const int wid  = tid >> 5, lane = tid & 31;
  const int wm   = wid & 3, wn = wid >> 2;       // 4x2 wave grid: 32 rows x 64 cols each
  const int half = lane >> 4, lrow = lane & 15;
  const int m0 = blockIdx.x * 128, n0 = blockIdx.y * 128;

  f32x8 acc[2][4];
  for (int i = 0; i < 2; ++i)
    for (int j = 0; j < 4; ++j) acc[i][j] = zero8();

  // async double-buffered staging: each thread owns half a 32-elem row (32B of A, 32B of B)
  const int ldr = tid >> 1, ldh = tid & 1;
  const unsigned short* ga0 = A  + (size_t)(m0 + ldr) * DM + ldh * 16;
  const unsigned short* gb0 = Bt + (size_t)(n0 + ldr) * DM + ldh * 16;
  unsigned la[2], lb[2];
  la[0] = lds_off(&As[0][ldr * 32 + ldh * 16]);
  la[1] = lds_off(&As[1][ldr * 32 + ldh * 16]);
  lb[0] = lds_off(&Bs[0][ldr * 32 + ldh * 16]);
  lb[1] = lds_off(&Bs[1][ldr * 32 + ldh * 16]);

  auto prefetch = [&](int buf, int k0) {   // 4 async b128 per wave-thread
    async_b128(la[buf],      ga0 + k0);
    async_b128(la[buf] + 16, ga0 + k0 + 8);
    async_b128(lb[buf],      gb0 + k0);
    async_b128(lb[buf] + 16, gb0 + k0 + 8);
  };

  prefetch(0, 0);
  for (int it = 0; it < 32; ++it) {
    const int cur = it & 1;
    if (it + 1 < 32) {
      prefetch(1 - cur, (it + 1) * 32);   // overlap next tile with current compute
      wait_async<4>();                    // in-order: first 4 (current tile) done
    } else {
      wait_async<0>();
    }
    __syncthreads();
    Frag af[2], bfv[4];
    for (int mt = 0; mt < 2; ++mt) {
      const unsigned short* p = &As[cur][(wm * 32 + mt * 16 + lrow) * 32];
      af[mt].lo = *(const uint4*)(p + half * 8);        // K {0..7}/{8..15}
      af[mt].hi = *(const uint4*)(p + 16 + half * 8);   // K {16..23}/{24..31}
    }
    for (int nt = 0; nt < 4; ++nt) {
      const unsigned short* p = &Bs[cur][(wn * 64 + nt * 16 + lrow) * 32 + half * 8];
      bfv[nt].lo = *(const uint4*)(p);                  // K {0..7}/{8..15}
      bfv[nt].hi = *(const uint4*)(p + 16);
    }
    for (int mt = 0; mt < 2; ++mt)
      for (int nt = 0; nt < 4; ++nt)
        acc[mt][nt] = wmma_bf16(af[mt], bfv[nt], acc[mt][nt]);
    __syncthreads();
  }

  for (int mt = 0; mt < 2; ++mt)
    for (int nt = 0; nt < 4; ++nt) {
      const int n = n0 + wn * 64 + nt * 16 + lrow;
      const int mbase = m0 + wm * 32 + mt * 16 + half * 8;
      const float b = bias[n];
      if (MODE == 0) {
        for (int r = 0; r < 8; ++r)
          outb[(size_t)(mbase + r) * DM + n] = f2bf(eluf(acc[mt][nt][r] + b));
      } else if (MODE == 1 || MODE == 2) {
        const int bI = mbase >> 12, nn = mbase & 4095;
        const int h = n >> 6, d = n & 63;
        unsigned short* dst = outb + ((size_t)((bI * NH + h) * HD + d)) * SEQ + nn;
        unsigned int pk[4];
        for (int r = 0; r < 4; ++r) {
          float v0 = acc[mt][nt][2 * r] + b, v1 = acc[mt][nt][2 * r + 1] + b;
          if (MODE == 1) { v0 = eluf(v0); v1 = eluf(v1); }
          pk[r] = (unsigned int)f2bf(v0) | ((unsigned int)f2bf(v1) << 16);
        }
        *(uint4*)dst = make_uint4(pk[0], pk[1], pk[2], pk[3]);
      } else {
        for (int r = 0; r < 8; ++r) {
          size_t idx = (size_t)(mbase + r) * DM + n;
          outf[idx] = acc[mt][nt][r] + b + xres[idx];
        }
      }
    }
}

// ---------------- Ksum[bh*64+d] = sum_n Kt[row][n] ----------------
__global__ __launch_bounds__(256) void k_ksum(const unsigned short* __restrict__ Kt,
                                              float* __restrict__ Ksum) {
  __shared__ float red[256];
  const unsigned short* p = Kt + (size_t)blockIdx.x * SEQ;
  float s = 0.f;
  for (int i = threadIdx.x; i < SEQ; i += 256) s += bf2f(p[i]);
  red[threadIdx.x] = s; __syncthreads();
  for (int off = 128; off > 0; off >>= 1) {
    if (threadIdx.x < off) red[threadIdx.x] += red[threadIdx.x + off];
    __syncthreads();
  }
  if (threadIdx.x == 0) Ksum[blockIdx.x] = red[0];
}

// ---------------- Z[bh*4096+n] = 1 / dot(Q[token, head], Ksum[bh]) ----------------
__global__ __launch_bounds__(256) void k_z(const unsigned short* __restrict__ Qb,
                                           const float* __restrict__ Ksum,
                                           float* __restrict__ Z) {
  int idx = blockIdx.x * 256 + threadIdx.x;   // 64*4096
  int bh = idx >> 12, n = idx & 4095;
  int b = bh >> 4, h = bh & 15;
  const unsigned short* q = Qb + (size_t)(b * SEQ + n) * DM + h * HD;
  const float* ks = Ksum + bh * HD;
  float s = 0.f;
  for (int d = 0; d < HD; ++d) s += bf2f(q[d]) * ks[d];
  Z[idx] = 1.0f / s;
}

__global__ __launch_bounds__(256) void k_zero(float* __restrict__ p, int n) {
  int i = blockIdx.x * 256 + threadIdx.x;
  if (i < n) p[i] = 0.f;
}

// ---------------- KVT[bh][e][d] += sum_n V[n,e] * K[n,d]  (== KV^T) ----------------
__global__ __launch_bounds__(128)
void k_kv(const unsigned short* __restrict__ Vt, const unsigned short* __restrict__ Kt,
          float* __restrict__ KVT) {
  const int bh = blockIdx.x, chunk = blockIdx.y;           // 64 x 8 (512-n chunks)
  const int wid = threadIdx.x >> 5, lane = threadIdx.x & 31;
  const int half = lane >> 4, lrow = lane & 15;
  f32x8 acc[4];
  for (int i = 0; i < 4; ++i) acc[i] = zero8();
  const unsigned short* Abase = Vt + ((size_t)bh * HD + wid * 16 + lrow) * SEQ;
  for (int ks = 0; ks < 16; ++ks) {
    const int nb = chunk * 512 + ks * 32;
    Frag a;
    a.lo = *(const uint4*)(Abase + nb + half * 8);
    a.hi = *(const uint4*)(Abase + nb + 16 + half * 8);
    for (int dt = 0; dt < 4; ++dt) {
      const unsigned short* bp = Kt + ((size_t)bh * HD + dt * 16 + lrow) * SEQ + nb + half * 8;
      Frag bf; bf.lo = *(const uint4*)bp; bf.hi = *(const uint4*)(bp + 16);
      acc[dt] = wmma_bf16(a, bf, acc[dt]);
    }
  }
  for (int dt = 0; dt < 4; ++dt) {
    const int e = wid * 16 + half * 8;
    const int d = dt * 16 + lrow;
    float* dst = KVT + ((size_t)bh * HD + e) * HD + d;
    for (int r = 0; r < 8; ++r) atomicAdd(dst + (size_t)r * HD, acc[dt][r]);
  }
}

__global__ __launch_bounds__(256) void k_kvpack(const float* __restrict__ KVT,
                                                unsigned short* __restrict__ KVTb) {
  int i = blockIdx.x * 256 + threadIdx.x;
  KVTb[i] = f2bf(KVT[i]);
}

// ---------------- attn[token, h*64+e] = Z[bh,n] * sum_d Q[token,d] KV[d,e] ----------------
__global__ __launch_bounds__(256)
void k_attn(const unsigned short* __restrict__ Qb, const unsigned short* __restrict__ KVTb,
            const float* __restrict__ Z, unsigned short* __restrict__ attnb) {
  __shared__ __align__(16) unsigned short Bs[HD * HD];   // whole KV^T head
  const int h = blockIdx.y;
  const int m0 = blockIdx.x * 128;
  const int bh = (m0 >> 12) * NH + h;
  const int tid = threadIdx.x, wid = tid >> 5, lane = tid & 31;
  const int half = lane >> 4, lrow = lane & 15;
  {
    const unsigned short* src = KVTb + (size_t)bh * HD * HD + tid * 16;
    unsigned l = lds_off(&Bs[tid * 16]);
    async_b128(l, src);
    async_b128(l + 16, src + 8);
    wait_async<0>();
  }
  __syncthreads();
  f32x8 acc[4];
  for (int i = 0; i < 4; ++i) acc[i] = zero8();
  const unsigned short* Arow = Qb + (size_t)(m0 + wid * 16 + lrow) * DM + h * HD;
  for (int ks = 0; ks < 2; ++ks) {
    Frag a;
    a.lo = *(const uint4*)(Arow + ks * 32 + half * 8);
    a.hi = *(const uint4*)(Arow + ks * 32 + 16 + half * 8);
    for (int nt = 0; nt < 4; ++nt) {
      const unsigned short* bp = Bs + (nt * 16 + lrow) * HD + ks * 32 + half * 8;
      Frag bf; bf.lo = *(const uint4*)bp; bf.hi = *(const uint4*)(bp + 16);
      acc[nt] = wmma_bf16(a, bf, acc[nt]);
    }
  }
  for (int nt = 0; nt < 4; ++nt) {
    const int e = nt * 16 + lrow;
    const int mb = m0 + wid * 16 + half * 8;
    for (int r = 0; r < 8; ++r) {
      const int m = mb + r;
      float z = Z[(size_t)bh * SEQ + (m & 4095)];
      attnb[(size_t)m * DM + h * HD + e] = f2bf(acc[nt][r] * z);
    }
  }
}

// ---------------- in-place LayerNorm over rows of y (d_out) ----------------
__global__ __launch_bounds__(256)
void k_ln(float* __restrict__ y, const float* __restrict__ gamma,
          const float* __restrict__ beta) {
  __shared__ float rs[256], rq[256];
  float* p = y + (size_t)blockIdx.x * DM;
  float v[4], s = 0.f, q = 0.f;
  for (int i = 0; i < 4; ++i) {
    v[i] = p[threadIdx.x + i * 256];
    s += v[i]; q += v[i] * v[i];
  }
  rs[threadIdx.x] = s; rq[threadIdx.x] = q; __syncthreads();
  for (int off = 128; off > 0; off >>= 1) {
    if (threadIdx.x < off) {
      rs[threadIdx.x] += rs[threadIdx.x + off];
      rq[threadIdx.x] += rq[threadIdx.x + off];
    }
    __syncthreads();
  }
  const float mu  = rs[0] * (1.f / DM);
  const float var = rq[0] * (1.f / DM) - mu * mu;
  const float inv = rsqrtf(var + 1e-3f);
  for (int i = 0; i < 4; ++i) {
    int c = threadIdx.x + i * 256;
    p[c] = (v[i] - mu) * inv * gamma[c] + beta[c];
  }
}

extern "C" void kernel_launch(void* const* d_in, const int* in_sizes, int n_in,
                              void* d_out, int out_size, void* d_ws, size_t ws_size,
                              hipStream_t stream) {
  (void)in_sizes; (void)n_in; (void)out_size; (void)ws_size;
  const float* x     = (const float*)d_in[0];
  const float* Wq    = (const float*)d_in[1];
  const float* bq    = (const float*)d_in[2];
  const float* Wk    = (const float*)d_in[3];
  const float* bk    = (const float*)d_in[4];
  const float* Wv    = (const float*)d_in[5];
  const float* bv    = (const float*)d_in[6];
  const float* Wo    = (const float*)d_in[7];
  const float* bo    = (const float*)d_in[8];
  const float* gamma = (const float*)d_in[9];
  const float* beta  = (const float*)d_in[10];
  float* out = (float*)d_out;

  char* ws = (char*)d_ws;
  size_t off = 0;
  auto alloc = [&](size_t bytes) -> void* {
    void* p = ws + off;
    off += (bytes + 255) & ~(size_t)255;
    return p;
  };
  unsigned short* xb    = (unsigned short*)alloc((size_t)TOK * DM * 2);
  unsigned short* Wt    = (unsigned short*)alloc((size_t)4 * DM * DM * 2);
  unsigned short* Qb    = (unsigned short*)alloc((size_t)TOK * DM * 2);
  unsigned short* Ktr   = (unsigned short*)alloc((size_t)TOK * DM * 2);
  unsigned short* Vtr   = (unsigned short*)alloc((size_t)TOK * DM * 2);
  unsigned short* attnb = (unsigned short*)alloc((size_t)TOK * DM * 2);
  float*          Ksum  = (float*)alloc((size_t)BATCH * NH * HD * 4);
  float*          Zbuf  = (float*)alloc((size_t)BATCH * NH * SEQ * 4);
  float*          KVT   = (float*)alloc((size_t)BATCH * NH * HD * HD * 4);
  unsigned short* KVTb  = (unsigned short*)alloc((size_t)BATCH * NH * HD * HD * 2);

  // 1) pack inputs to bf16 (weights transposed)
  k_pack_x<<<TOK * DM / 1024, 256, 0, stream>>>(x, xb);
  k_pack_wt<<<dim3(DM * DM / 256, 4), 256, 0, stream>>>(Wq, Wk, Wv, Wo, Wt);

  // 2) QKV projections (WMMA GEMMs, async-to-LDS double buffering)
  dim3 gg(TOK / 128, DM / 128);
  k_gemm<0><<<gg, 256, 0, stream>>>(xb, Wt,               bq, nullptr, Qb,  nullptr);
  k_gemm<1><<<gg, 256, 0, stream>>>(xb, Wt + DM * DM,     bk, nullptr, Ktr, nullptr);
  k_gemm<2><<<gg, 256, 0, stream>>>(xb, Wt + 2 * DM * DM, bv, nullptr, Vtr, nullptr);

  // 3) Ksum, Z
  k_ksum<<<BATCH * NH * HD, 256, 0, stream>>>(Ktr, Ksum);
  k_z<<<BATCH * NH * SEQ / 256, 256, 0, stream>>>(Qb, Ksum, Zbuf);

  // 4) KV^T = sum_n V K (WMMA, split-N with f32 atomics)
  k_zero<<<(BATCH * NH * HD * HD + 255) / 256, 256, 0, stream>>>(KVT, BATCH * NH * HD * HD);
  k_kv<<<dim3(BATCH * NH, 8), 128, 0, stream>>>(Vtr, Ktr, KVT);
  k_kvpack<<<BATCH * NH * HD * HD / 256, 256, 0, stream>>>(KVT, KVTb);

  // 5) attn = Z * (Q @ KV) (WMMA, B-head resident in LDS via async copy)
  k_attn<<<dim3(TOK / 128, NH), 256, 0, stream>>>(Qb, KVTb, Zbuf, attnb);

  // 6) output projection + bias + residual (WMMA), then in-place LayerNorm
  k_gemm<3><<<gg, 256, 0, stream>>>(attnb, Wt + 3 * DM * DM, bo, x, nullptr, out);
  k_ln<<<TOK, 256, 0, stream>>>(out, gamma, beta);
}
